// MoELayer_86406152061623
// MI455X (gfx1250) — matrix operand order repaired and checked by reading, not compiled, not run on hardware
//
#include <hip/hip_runtime.h>
#include <math.h>

// ---------------- problem constants ----------------
#define HID   512
#define INTER 2048
#define NEXP  8
#define TOPK  2
#define NTOK  4096            // BATCH*SEQ
#define MTILE 32
#define MAXTILES (NTOK * TOPK / MTILE + NEXP)   // 264

// ---------------- WMMA types ----------------
typedef __attribute__((ext_vector_type(16))) __bf16 v16bf;
typedef __attribute__((ext_vector_type(8)))  float  v8f;

union Frag16 { v16bf v; unsigned u[8]; uint4 q[2]; };

// round-to-nearest-even f32 -> bf16 bits
static __device__ inline unsigned f2bf(float f) {
    unsigned u = __float_as_uint(f);
    return (u + 0x7FFFu + ((u >> 16) & 1u)) >> 16;
}
static __device__ inline unsigned pack2bf(float lo, float hi) {
    return f2bf(lo) | (f2bf(hi) << 16);
}

// branchless erf (Abramowitz-Stegun 7.1.26, |err| < 1.5e-7), no EXEC divergence
static __device__ inline float erf_branchless(float z) {
    float az = fabsf(z);
    float t  = __builtin_amdgcn_rcpf(fmaf(0.3275911f, az, 1.0f));
    float p  = fmaf(1.061405429f, t, -1.453152027f);
    p = fmaf(p, t, 1.421413741f);
    p = fmaf(p, t, -0.284496736f);
    p = fmaf(p, t, 0.254829592f);
    p = p * t;
    float y = fmaf(-p, __expf(-az * az), 1.0f);
    return copysignf(y, z);
}
static __device__ inline float gelu_exact(float v) {
    return 0.5f * v * (1.0f + erf_branchless(v * 0.70710678118654752f));
}

// ---------------- kernel 0: f32 -> packed bf16 weight conversion ----------------
__global__ void conv_bf16_kernel(const float* __restrict__ src,
                                 unsigned* __restrict__ dst, int npairs) {
    int i = blockIdx.x * blockDim.x + threadIdx.x;
    int stride = gridDim.x * blockDim.x;
    for (int j = i; j < npairs; j += stride) {
        float2 f = ((const float2*)src)[j];
        dst[j] = pack2bf(f.x, f.y);
    }
}

// ---------------- kernel 1: init ----------------
__global__ void moe_init_kernel(float* __restrict__ out, int* __restrict__ list,
                                int* __restrict__ cursor, int* __restrict__ counts) {
    int i = blockIdx.x * blockDim.x + threadIdx.x;
    int stride = gridDim.x * blockDim.x;
    for (int j = i; j < NTOK * HID; j += stride) out[j] = 0.0f;
    if (i < MAXTILES * MTILE) list[i] = -1;
    if (i < NEXP) { cursor[i] = 0; counts[i] = 0; }
}

// ---------------- kernel 2: router ----------------
__global__ __launch_bounds__(256)
void moe_router_kernel(const float* __restrict__ x, const float* __restrict__ wr,
                       int* __restrict__ top_idx, float* __restrict__ top_w,
                       int* __restrict__ counts, float* __restrict__ partial) {
    __shared__ float s_wr[HID * NEXP];     // 16 KB
    __shared__ float s_probs[256 * NEXP];  //  8 KB
    int tid = threadIdx.x;
    for (int i = tid; i < HID * NEXP; i += 256) s_wr[i] = wr[i];
    __syncthreads();

    int t = blockIdx.x * 256 + tid;
    float acc[NEXP];
#pragma unroll
    for (int e = 0; e < NEXP; ++e) acc[e] = 0.0f;
    const float* xr = x + (size_t)t * HID;
    for (int h = 0; h < HID; ++h) {
        float xv = xr[h];
#pragma unroll
        for (int e = 0; e < NEXP; ++e) acc[e] += xv * s_wr[h * NEXP + e];
    }
    // top-2 (ties -> lowest index, matching jax.lax.top_k)
    int i0 = 0; float v0 = acc[0];
#pragma unroll
    for (int e = 1; e < NEXP; ++e) if (acc[e] > v0) { v0 = acc[e]; i0 = e; }
    int i1 = -1; float v1 = -3.4e38f;
#pragma unroll
    for (int e = 0; e < NEXP; ++e) if (e != i0 && acc[e] > v1) { v1 = acc[e]; i1 = e; }
    // softmax over the two selected logits (v0 >= v1)
    float e1 = __expf(v1 - v0);
    float inv2 = 1.0f / (1.0f + e1);
    top_idx[t * 2]     = i0;  top_w[t * 2]     = inv2;
    top_idx[t * 2 + 1] = i1;  top_w[t * 2 + 1] = e1 * inv2;
    atomicAdd(&counts[i0], 1);
    atomicAdd(&counts[i1], 1);

    // full softmax for usage stats
    float s = 0.0f, p[NEXP];
#pragma unroll
    for (int e = 0; e < NEXP; ++e) { p[e] = __expf(acc[e] - v0); s += p[e]; }
    float invs = 1.0f / s;
#pragma unroll
    for (int e = 0; e < NEXP; ++e) s_probs[tid * NEXP + e] = p[e] * invs;
    __syncthreads();
    // fixed-order tree reduction -> deterministic
    for (int strd = 128; strd > 0; strd >>= 1) {
        if (tid < strd) {
#pragma unroll
            for (int e = 0; e < NEXP; ++e)
                s_probs[tid * NEXP + e] += s_probs[(tid + strd) * NEXP + e];
        }
        __syncthreads();
    }
    if (tid < NEXP) partial[blockIdx.x * NEXP + tid] = s_probs[tid];
}

// ---------------- kernel 3: finalize router ----------------
__global__ void moe_finalize_kernel(const int* __restrict__ counts,
                                    const float* __restrict__ partial,
                                    float* __restrict__ out_tail,   // d_out + NTOK*HID
                                    int* __restrict__ tile_starts,
                                    int* __restrict__ total_tiles) {
    __shared__ float usage[NEXP];
    int tid = threadIdx.x;
    if (tid < NEXP) {
        float s = 0.0f;
        for (int b = 0; b < 16; ++b) s += partial[b * NEXP + tid];
        float u = s / (float)NTOK;
        usage[tid] = u;
        out_tail[1 + tid] = u;            // expert_usage
    }
    __syncthreads();
    if (tid == 0) {
        float loss = 0.0f;
        for (int e = 0; e < NEXP; ++e) {
            float d = usage[e] - 1.0f / (float)NEXP;
            loss += d * d;
        }
        out_tail[0] = loss;               // load_balance_loss
        int cum = 0;
        for (int e = 0; e < NEXP; ++e) {
            tile_starts[e] = cum;
            cum += (counts[e] + MTILE - 1) / MTILE;
        }
        tile_starts[NEXP] = cum;
        *total_tiles = cum;
    }
}

// ---------------- kernel 4: scatter token list ----------------
__global__ void moe_scatter_kernel(const int* __restrict__ top_idx,
                                   const int* __restrict__ tile_starts,
                                   int* __restrict__ cursor, int* __restrict__ list) {
    int i = blockIdx.x * blockDim.x + threadIdx.x;   // 0 .. NTOK*TOPK-1
    int e = top_idx[i];
    int pos = atomicAdd(&cursor[e], 1);
    list[tile_starts[e] * MTILE + pos] = i;          // entry encodes (token<<1)|slot
}

// ---------------- kernel 5: grouped FFN with WMMA (2x2 register tile) ----------------
__global__ __launch_bounds__(256)
void moe_ffn_kernel(const float* __restrict__ x,
                    const unsigned short* __restrict__ wu_bf,   // (E, HID, INTER) bf16
                    const unsigned short* __restrict__ wd_bf,   // (E, INTER, HID) bf16
                    const int* __restrict__ list,
                    const float* __restrict__ top_w,
                    const int* __restrict__ tile_starts,
                    const int* __restrict__ total_tiles,
                    float* __restrict__ out) {
    extern __shared__ char smem[];
    unsigned short* Xs = (unsigned short*)smem;                        // 32x512 bf16 = 32 KB
    unsigned short* Hs = (unsigned short*)(smem + MTILE * HID * 2);    // 32x2048 bf16 = 128 KB
    int*   s_tok = (int*)(smem + MTILE * HID * 2 + MTILE * INTER * 2);
    float* s_wgt = (float*)(s_tok + MTILE);

    int b = blockIdx.x;
    if (b >= *total_tiles) return;

    // which expert owns this tile
    int e = 0;
#pragma unroll
    for (int i = 1; i < NEXP; ++i) if (b >= tile_starts[i]) e = i;

    int tid = threadIdx.x;

    // ---- load & convert X tile (gathered) ----
    {
        int r  = tid >> 3;            // 0..31 row
        int c0 = (tid & 7) * 64;      // 64 floats per thread
        int entry = list[b * MTILE + r];
        if ((tid & 7) == 0) {
            s_tok[r] = (entry >= 0) ? (entry >> 1) : -1;
            s_wgt[r] = (entry >= 0) ? top_w[entry] : 0.0f;
        }
        unsigned* dst = (unsigned*)&Xs[r * HID + c0];
        if (entry >= 0) {
            const float4* xr4 = (const float4*)(x + (size_t)(entry >> 1) * HID + c0);
#pragma unroll
            for (int j = 0; j < 16; ++j) {
                float4 f = xr4[j];
                dst[2 * j]     = pack2bf(f.x, f.y);
                dst[2 * j + 1] = pack2bf(f.z, f.w);
            }
        } else {
#pragma unroll
            for (int j = 0; j < 32; ++j) dst[j] = 0u;
        }
    }
    __syncthreads();

    int lane  = tid & 31;
    int wv    = tid >> 5;            // wave 0..7
    int row16 = lane & 15;           // A row / C column within tile
    int khalf = (lane >> 4) * 8;     // A-fragment K half select
    int crow  = (lane >> 4) * 8;     // C/D row base: lanes 16-31 -> M=8..15

    // ---- up-projection: H = gelu(X @ Wu); 128 N-tiles, pairs per wave ----
    const unsigned short* Wu = wu_bf + (size_t)e * HID * INTER;
    for (int nt = wv * 2; nt < INTER / 16; nt += 16) {
        int n0a = nt * 16, n0b = n0a + 16;
        v8f acc00 = {}, acc01 = {}, acc10 = {}, acc11 = {};
        for (int kb = 0; kb < HID; kb += 32) {
            // B fragments (bf16 gathered): lane = N column, elements = K in table order
            const unsigned short* pb0 = Wu + (size_t)(kb + khalf) * INTER + n0a + row16;
            Frag16 b0, b1;
#pragma unroll
            for (int j = 0; j < 8; ++j) {
                int ko = (j < 4) ? 2 * j : 2 * j + 8;
                b0.u[j] = (unsigned)pb0[ko * INTER] | ((unsigned)pb0[(ko + 1) * INTER] << 16);
                b1.u[j] = (unsigned)pb0[ko * INTER + 16] | ((unsigned)pb0[(ko + 1) * INTER + 16] << 16);
            }
            // A fragments from LDS: two contiguous 16B runs at +0 / +32 bytes
            const uint4* pa0 = (const uint4*)&Xs[row16 * HID + kb + khalf];
            const uint4* pa1 = (const uint4*)&Xs[(row16 + 16) * HID + kb + khalf];
            Frag16 a0, a1;
            a0.q[0] = pa0[0]; a0.q[1] = pa0[2];
            a1.q[0] = pa1[0]; a1.q[1] = pa1[2];
            if (kb + 32 < HID)
                __builtin_prefetch(Wu + (size_t)(kb + 32 + lane) * INTER + n0a, 0, 1);
            acc00 = __builtin_amdgcn_wmma_f32_16x16x32_bf16(false, a0.v, false, b0.v,
                                                            (short)0, acc00, false, false);
            acc01 = __builtin_amdgcn_wmma_f32_16x16x32_bf16(false, a0.v, false, b1.v,
                                                            (short)0, acc01, false, false);
            acc10 = __builtin_amdgcn_wmma_f32_16x16x32_bf16(false, a1.v, false, b0.v,
                                                            (short)0, acc10, false, false);
            acc11 = __builtin_amdgcn_wmma_f32_16x16x32_bf16(false, a1.v, false, b1.v,
                                                            (short)0, acc11, false, false);
        }
        // branchless exact-erf GELU, convert to bf16, stage into LDS
#pragma unroll
        for (int v = 0; v < 8; ++v) {
            int rr = v + crow;
            Hs[rr * INTER + n0a + row16]        = (unsigned short)f2bf(gelu_exact(acc00[v]));
            Hs[rr * INTER + n0b + row16]        = (unsigned short)f2bf(gelu_exact(acc01[v]));
            Hs[(rr + 16) * INTER + n0a + row16] = (unsigned short)f2bf(gelu_exact(acc10[v]));
            Hs[(rr + 16) * INTER + n0b + row16] = (unsigned short)f2bf(gelu_exact(acc11[v]));
        }
    }
    __syncthreads();

    // ---- down-projection: Y = H @ Wd; 32 N-tiles, pairs per wave ----
    const unsigned short* Wd = wd_bf + (size_t)e * INTER * HID;
    for (int nt = wv * 2; nt < HID / 16; nt += 16) {
        int n0a = nt * 16, n0b = n0a + 16;
        v8f acc00 = {}, acc01 = {}, acc10 = {}, acc11 = {};
        for (int kb = 0; kb < INTER; kb += 32) {
            const unsigned short* pb0 = Wd + (size_t)(kb + khalf) * HID + n0a + row16;
            Frag16 b0, b1;
#pragma unroll
            for (int j = 0; j < 8; ++j) {
                int ko = (j < 4) ? 2 * j : 2 * j + 8;
                b0.u[j] = (unsigned)pb0[ko * HID] | ((unsigned)pb0[(ko + 1) * HID] << 16);
                b1.u[j] = (unsigned)pb0[ko * HID + 16] | ((unsigned)pb0[(ko + 1) * HID + 16] << 16);
            }
            const uint4* pa0 = (const uint4*)&Hs[row16 * INTER + kb + khalf];
            const uint4* pa1 = (const uint4*)&Hs[(row16 + 16) * INTER + kb + khalf];
            Frag16 a0, a1;
            a0.q[0] = pa0[0]; a0.q[1] = pa0[2];
            a1.q[0] = pa1[0]; a1.q[1] = pa1[2];
            if (kb + 32 < INTER)
                __builtin_prefetch(Wd + (size_t)(kb + 32 + lane) * HID + n0a, 0, 1);
            acc00 = __builtin_amdgcn_wmma_f32_16x16x32_bf16(false, a0.v, false, b0.v,
                                                            (short)0, acc00, false, false);
            acc01 = __builtin_amdgcn_wmma_f32_16x16x32_bf16(false, a0.v, false, b1.v,
                                                            (short)0, acc01, false, false);
            acc10 = __builtin_amdgcn_wmma_f32_16x16x32_bf16(false, a1.v, false, b0.v,
                                                            (short)0, acc10, false, false);
            acc11 = __builtin_amdgcn_wmma_f32_16x16x32_bf16(false, a1.v, false, b1.v,
                                                            (short)0, acc11, false, false);
        }
        // weighted scatter-add: each output element gets exactly 2 commutative adds
#pragma unroll
        for (int v = 0; v < 8; ++v) {
            int rr = v + crow;
            int t0 = s_tok[rr];
            int t1 = s_tok[rr + 16];
            if (t0 >= 0) {
                atomicAdd(&out[(size_t)t0 * HID + n0a + row16], acc00[v] * s_wgt[rr]);
                atomicAdd(&out[(size_t)t0 * HID + n0b + row16], acc01[v] * s_wgt[rr]);
            }
            if (t1 >= 0) {
                atomicAdd(&out[(size_t)t1 * HID + n0a + row16], acc10[v] * s_wgt[rr + 16]);
                atomicAdd(&out[(size_t)t1 * HID + n0b + row16], acc11[v] * s_wgt[rr + 16]);
            }
        }
    }
}

// ---------------- host-side launcher ----------------
extern "C" void kernel_launch(void* const* d_in, const int* in_sizes, int n_in,
                              void* d_out, int out_size, void* d_ws, size_t ws_size,
                              hipStream_t stream) {
    const float* x  = (const float*)d_in[0];   // (4096, 512)
    const float* wr = (const float*)d_in[1];   // (512, 8)
    const float* wu = (const float*)d_in[2];   // (8, 512, 2048)
    const float* wd = (const float*)d_in[3];   // (8, 2048, 512)
    float* out = (float*)d_out;                // 4096*512 + 1 + 8

    // workspace layout: small control arrays first, then bf16 weight mirrors (~33.7 MB)
    char* p = (char*)d_ws;
    int*   top_idx     = (int*)p;   p += NTOK * TOPK * sizeof(int);
    float* top_w       = (float*)p; p += NTOK * TOPK * sizeof(float);
    int*   counts      = (int*)p;   p += NEXP * sizeof(int);
    int*   cursor      = (int*)p;   p += NEXP * sizeof(int);
    int*   tile_starts = (int*)p;   p += (NEXP + 1) * sizeof(int);
    int*   total_tiles = (int*)p;   p += 4;
    float* partial     = (float*)p; p += 16 * NEXP * sizeof(float);
    int*   list        = (int*)p;   p += MAXTILES * MTILE * sizeof(int);
    p = (char*)(((uintptr_t)p + 255) & ~(uintptr_t)255);
    unsigned short* wu_bf = (unsigned short*)p; p += (size_t)NEXP * HID * INTER * 2;
    unsigned short* wd_bf = (unsigned short*)p; p += (size_t)NEXP * INTER * HID * 2;

    const int wpairs = NEXP * HID * INTER / 2;   // 4,194,304 dword-pairs each
    conv_bf16_kernel<<<2048, 256, 0, stream>>>(wu, (unsigned*)wu_bf, wpairs);
    conv_bf16_kernel<<<2048, 256, 0, stream>>>(wd, (unsigned*)wd_bf, wpairs);

    moe_init_kernel<<<1024, 256, 0, stream>>>(out, list, cursor, counts);
    moe_router_kernel<<<NTOK / 256, 256, 0, stream>>>(x, wr, top_idx, top_w, counts, partial);
    moe_finalize_kernel<<<1, 32, 0, stream>>>(counts, partial, out + (size_t)NTOK * HID,
                                              tile_starts, total_tiles);
    moe_scatter_kernel<<<(NTOK * TOPK) / 256, 256, 0, stream>>>(top_idx, tile_starts, cursor, list);

    size_t shmem = (size_t)MTILE * HID * 2 + (size_t)MTILE * INTER * 2
                 + MTILE * sizeof(int) + MTILE * sizeof(float);   // ~160.25 KB
    moe_ffn_kernel<<<MAXTILES, 256, shmem, stream>>>(x, wu_bf, wd_bf, list, top_w,
                                                     tile_starts, total_tiles, out);
}
